// OpenUnmixCDAE_B_34205119546051
// MI455X (gfx1250) — matrix operand order repaired and checked by reading, not compiled, not run on hardware
//
#include <hip/hip_runtime.h>
#include <hip/hip_bf16.h>

typedef __attribute__((ext_vector_type(16))) _Float16 v16h;
typedef __attribute__((ext_vector_type(8)))  float    v8f;

#define DEV __device__ __forceinline__

static constexpr int B_ = 2, C_ = 2, S_ = 128, TOTF = 210, MAXBIN = 180;
static constexpr int MAXT = 8256;
static constexpr int KHH = 3, KWW = 5;
static constexpr float EPSV = 1e-5f;

// ---------------------------------------------------------------------------
// A-fragment K index for 16-bit WMMA A (16x32): lanes 0-15 hold K0-15,
// lanes 16-31 hold K16-31 (contiguous, matching the B fragments).
DEV int kidxA(int lane, int j) { return j + ((lane >= 16) ? 16 : 0); }

// ---------------------------------------------------------------------------
// Kernel 1: overlap-add + repeat-interpolation + input normalization (fused).
// mix stays f32 (feeds final masking); encoder input is stored f16.
__global__ void ola_interp_kernel(const float* __restrict__ x,
                                  float* __restrict__ mix,
                                  _Float16* __restrict__ enc_in,
                                  const float* __restrict__ mean,
                                  const float* __restrict__ scale,
                                  int f, int w, int F0, int rep)
{
    long long e = (long long)blockIdx.x * blockDim.x + threadIdx.x;
    long long total = (long long)B_ * C_ * f * MAXT;
    if (e >= total) return;
    int t  = (int)(e % MAXT);
    long long r = e / MAXT;
    int fl = (int)(r % f); r /= f;
    int c  = (int)(r % C_);
    int b  = (int)(r / C_);
    int h  = w >> 1;
    int tt = t / rep;
    int s  = tt / h;
    int j  = tt - s * h;
    const float* xb = x + ((long long)(b * C_ + c) * f + fl) * S_ * (long long)w;
    float val = 0.f;
    if (s < S_) val += xb[(long long)s * w + j];
    if (s >= 1) val += xb[(long long)(s - 1) * w + h + j];
    int frg = F0 + fl;
    mix[((long long)(b * C_ + c) * TOTF + frg) * MAXT + t] = val;
    if (frg < MAXBIN)
        enc_in[((long long)(b * C_ + c) * MAXBIN + frg) * MAXT + t] =
            (_Float16)((val + mean[frg]) * scale[frg]);
}

// ---------------------------------------------------------------------------
// Pre-swizzle conv weights (f32, OIHW or transposed+flipped IOHW) into the
// per-lane v16h A-fragment layout, converted to f16.
__global__ void prep_weights_kernel(const float* __restrict__ w,
                                    _Float16* __restrict__ dst,
                                    int Cout, int Cin, int MT, int KT, int transposed)
{
    int e = blockIdx.x * blockDim.x + threadIdx.x;
    int total = KT * MT * 512;
    if (e >= total) return;
    int j    = e & 15;
    int lane = (e >> 4) & 31;
    int mt   = (e >> 9) % MT;
    int kc   = e / (512 * MT);
    int m = mt * 16 + (lane & 15);
    int k = kc * 32 + kidxA(lane, j);
    int Kreal = Cin * KHH * KWW;
    float val = 0.f;
    if (m < Cout && k < Kreal) {
        int ci  = k / (KHH * KWW);
        int rem = k % (KHH * KWW);
        int kh = rem / KWW, kw = rem % KWW;
        if (!transposed)
            val = w[((m * Cin + ci) * KHH + kh) * KWW + kw];
        else  // convT: wf[co][ci][kh][kw] = w[ci][co][2-kh][4-kw]
            val = w[((ci * Cout + m) * KHH + (KHH - 1 - kh)) * KWW + (KWW - 1 - kw)];
    }
    dst[e] = (_Float16)val;
}

// ---------------------------------------------------------------------------
// Pre-swizzle deov weights: f16 copy with padded k-stride WS (multiple of 32B)
// so a B fragment is one aligned contiguous v16h load: B[k][v] = W[v][k].
__global__ void prep_deovw_kernel(const float* __restrict__ w,
                                  _Float16* __restrict__ dst, int W, int WS)
{
    int e = blockIdx.x * blockDim.x + threadIdx.x;
    int total = W * WS;
    if (e >= total) return;
    int v = e / WS, k = e - v * WS;
    dst[e] = (k < W) ? (_Float16)w[v * W + k] : (_Float16)0.f;
}

// ---------------------------------------------------------------------------
// Implicit-GEMM conv (+fused BN/ReLU or bias/sigmoid) with v_wmma_f32_16x16x32_f16.
// f16 in / f16 out activations (memory-bound pipeline: halves HBM traffic).
// M = Cout (MT m-tiles), N = 64 output columns per block (1 n-tile/wave),
// K = Cin*15 in chunks of 32, double-buffered LDS B tiles (1 barrier/chunk).
#define WTILE 64
#define BSTRIDE 48  // halves per column in LDS B tile (96B: keeps v16h 32B-aligned)

template <int MT, int MODE>
__global__ __launch_bounds__(128)
void conv_wmma_kernel(const _Float16* __restrict__ in, _Float16* __restrict__ out,
                      const _Float16* __restrict__ wf16,
                      const float* __restrict__ p0, const float* __restrict__ p1,
                      const float* __restrict__ p2, const float* __restrict__ p3,
                      int Cin, int Cout, int Hin, int Win, int Hout, int Wout,
                      int KT, int ph, int pw)
{
    __shared__ __align__(32) _Float16 Bl[2][WTILE * BSTRIDE];
    int tid  = threadIdx.x;
    int lane = tid & 31;
    int wv   = tid >> 5;
    int x0   = blockIdx.x * WTILE;
    int y    = blockIdx.y;
    int n    = blockIdx.z;
    int Kreal = Cin * KHH * KWW;
    const long long inN = (long long)n * Cin * Hin * Win;
    const long long chan2 = 2LL * Hin * Win;   // +2 input channels ahead

    auto stage = [&](int kc, _Float16* dstB) {
        for (int e = tid; e < 32 * WTILE; e += 128) {
            int col = e & (WTILE - 1);
            int kl  = e >> 6;
            int k   = kc * 32 + kl;
            _Float16 val = (_Float16)0.f;
            if (k < Kreal) {
                int ci  = k / (KHH * KWW);
                int rem = k % (KHH * KWW);
                int kh = rem / KWW, kw = rem % KWW;
                int iy = y + kh - ph;
                int ix = x0 + col + kw - pw;
                if (iy >= 0 && iy < Hin && ix >= 0 && ix < Win) {
                    long long gi = inN + ((long long)ci * Hin + iy) * Win + ix;
                    val = in[gi];
                    // pull the lines two channels (~one K-chunk) ahead toward WGP
                    if (e == tid && ci + 2 < Cin)
                        __builtin_prefetch(in + gi + chan2, 0, 3);
                }
            }
            dstB[col * BSTRIDE + kl] = val;
        }
    };

    v8f acc[MT] = {};
    stage(0, Bl[0]);
    __syncthreads();

    for (int kc = 0; kc < KT; ++kc) {
        const _Float16* cur = Bl[kc & 1];
        int col = wv * 16 + (lane & 15);
        v16h bf = *(const v16h*)&cur[col * BSTRIDE + ((lane >= 16) ? 16 : 0)];
#pragma unroll
        for (int mt = 0; mt < MT; ++mt) {
            v16h af = *(const v16h*)(wf16 + (((long long)kc * MT + mt) * 32 + lane) * 16);
            acc[mt] = __builtin_amdgcn_wmma_f32_16x16x32_f16(
                false, af, false, bf, (short)0, acc[mt], false, false);
        }
        if (kc + 1 < KT) stage(kc + 1, Bl[(kc + 1) & 1]);
        __syncthreads();
    }

    int colx = x0 + wv * 16 + (lane & 15);
    if (colx < Wout) {
#pragma unroll
        for (int mt = 0; mt < MT; ++mt) {
#pragma unroll
            for (int r = 0; r < 8; ++r) {
                int m = mt * 16 + r + ((lane >= 16) ? 8 : 0);
                if (m < Cout) {
                    float val = acc[mt][r];
                    if (MODE == 0) {            // BN + ReLU
                        float sc = p0[m] * rsqrtf(p3[m] + EPSV);
                        val = (val - p2[m]) * sc + p1[m];
                        val = fmaxf(val, 0.f);
                    } else {                    // bias + sigmoid
                        val = val + p0[m];
                        val = 1.f / (1.f + __expf(-val));
                    }
                    out[(((long long)n * Cout + m) * Hout + y) * Wout + colx] = (_Float16)val;
                }
            }
        }
    }
}

// ---------------------------------------------------------------------------
// Final stage, templated on W so all fragment loads are aligned 32B vectors:
// mask (sigmoid-head * mix, or passthrough-normalized * mix), stride-subsample,
// and the per-frequency (128 x W)·(W x W) einsum via WMMA.
// Key identity: wins[s][j] = blk[s*h + j] (overlapping contiguous slices).
template <int W>
__global__ __launch_bounds__(128)
void deov_kernel(const _Float16* __restrict__ tbuf, const float* __restrict__ mix,
                 const float* __restrict__ mean, const float* __restrict__ scale,
                 const _Float16* __restrict__ Wf, const float* __restrict__ bias,
                 float* __restrict__ out, int f, int F0, int rep)
{
    constexpr int H     = W / 2;
    constexpr int NCOEF = (S_ + 1) * H;
    constexpr int KC    = (W + 31) / 32;
    constexpr int NT    = W / 16;
    constexpr int WS    = (W < 32) ? 32 : W;   // padded k-stride of prepped weights
    __shared__ __align__(32) _Float16 sU[NCOEF];       // masked coefficients
    __shared__ __align__(32) _Float16 sA[8 * KC * 512];// fragment-ready A tiles

    int tid  = threadIdx.x;
    int lane = tid & 31;
    int wvv  = tid >> 5;
    int fl = blockIdx.x, c = blockIdx.y, b = blockIdx.z;
    int frg = F0 + fl;
    long long mixbase = ((long long)(b * C_ + c) * TOTF + frg) * MAXT;
    long long tbase   = ((long long)(b * C_ + c) * MAXBIN + frg) * MAXT;

    // 1) stage masked, subsampled coefficient vector
    for (int idx = tid; idx < NCOEF; idx += 128) {
        int t = idx * rep;
        float mv = mix[mixbase + t];
        float xm;
        if (frg < MAXBIN) xm = (float)tbuf[tbase + t] * mv;
        else              xm = (mv + mean[frg]) * scale[frg] * mv;
        sU[idx] = (_Float16)xm;
    }
    __syncthreads();

    // 2) build fragment-ready A tiles (unconditional LDS read + select for pad)
    for (int e = tid; e < 8 * KC * 512; e += 128) {
        int j     = e & 15;
        int lane2 = (e >> 4) & 31;
        int mtkc  = e >> 9;            // == mt*KC + kc
        int mt = mtkc / KC, kc = mtkc - mt * KC;
        int s = mt * 16 + (lane2 & 15);
        int k = kc * 32 + kidxA(lane2, j);
        int kk = (k < W) ? k : 0;
        _Float16 v = sU[s * H + kk];
        if (k >= W) v = (_Float16)0.f;
        sA[e] = v;
    }
    __syncthreads();

    // 3) 8 x NT WMMA tiles split over 4 waves; B fragments straight from L2
    long long obase = ((long long)(b * C_ + c) * f + fl) * S_ * (long long)W;
    for (int tl = wvv; tl < 8 * NT; tl += 4) {
        int mt = tl / NT, nt = tl - mt * NT;
        int vcol = nt * 16 + (lane & 15);
        v8f acc = {};
#pragma unroll
        for (int kc = 0; kc < KC; ++kc) {
            v16h af = *(const v16h*)(sA + ((mt * KC + kc) << 9) + lane * 16);
            v16h bf = *(const v16h*)(Wf + vcol * WS + kc * 32 + ((lane >= 16) ? 16 : 0));
            acc = __builtin_amdgcn_wmma_f32_16x16x32_f16(
                false, af, false, bf, (short)0, acc, false, false);
        }
        float bb = bias[vcol];
#pragma unroll
        for (int r = 0; r < 8; ++r) {
            int srow = mt * 16 + r + ((lane >= 16) ? 8 : 0);
            out[obase + (long long)srow * W + vcol] = acc[r] + bb;
        }
    }
}

// ---------------------------------------------------------------------------
extern "C" void kernel_launch(void* const* d_in, const int* in_sizes, int n_in,
                              void* d_out, int out_size, void* d_ws, size_t ws_size,
                              hipStream_t stream)
{
    (void)in_sizes; (void)n_in; (void)out_size; (void)ws_size;

    const float* xin[4] = { (const float*)d_in[0], (const float*)d_in[1],
                            (const float*)d_in[2], (const float*)d_in[3] };
    const float* mean  = (const float*)d_in[4];
    const float* scale = (const float*)d_in[5];
    const float* enc_w[3] = { (const float*)d_in[6],  (const float*)d_in[11], (const float*)d_in[16] };
    const float* enc_g[3] = { (const float*)d_in[7],  (const float*)d_in[12], (const float*)d_in[17] };
    const float* enc_b[3] = { (const float*)d_in[8],  (const float*)d_in[13], (const float*)d_in[18] };
    const float* enc_m[3] = { (const float*)d_in[9],  (const float*)d_in[14], (const float*)d_in[19] };
    const float* enc_v[3] = { (const float*)d_in[10], (const float*)d_in[15], (const float*)d_in[20] };
    const float* dec_w[2] = { (const float*)d_in[21], (const float*)d_in[26] };
    const float* dec_g[2] = { (const float*)d_in[22], (const float*)d_in[27] };
    const float* dec_b[2] = { (const float*)d_in[23], (const float*)d_in[28] };
    const float* dec_m[2] = { (const float*)d_in[24], (const float*)d_in[29] };
    const float* dec_v[2] = { (const float*)d_in[25], (const float*)d_in[30] };
    const float* dec2_w    = (const float*)d_in[31];
    const float* dec2_bias = (const float*)d_in[32];
    const float* deov_w[4] = { (const float*)d_in[33], (const float*)d_in[35],
                               (const float*)d_in[37], (const float*)d_in[39] };
    const float* deov_b[4] = { (const float*)d_in[34], (const float*)d_in[36],
                               (const float*)d_in[38], (const float*)d_in[40] };
    float* out = (float*)d_out;

    // Workspace layout (bytes): mix f32 | bufA f16 | bufB f16 | f16 weights
    char* ws = (char*)d_ws;
    float*    mix  = (float*)ws;                               // 6,935,040 f32
    _Float16* bufA = (_Float16*)(ws + 27740160ULL);            // 137,707,776 f16 max
    _Float16* bufB = (_Float16*)(ws + 303155712ULL);
    _Float16* wf   = (_Float16*)(ws + 578571264ULL);
    _Float16* wfE0 = wf +     0;  // KT=1,  MT=1 ->   512 halves
    _Float16* wfE1 = wf +   512;  // KT=6,  MT=2 ->  6144
    _Float16* wfE2 = wf +  6656;  // KT=12, MT=3 -> 18432
    _Float16* wfD0 = wf + 25088;  // KT=23, MT=2 -> 23552
    _Float16* wfD1 = wf + 48640;  // KT=12, MT=1 ->  6144
    _Float16* wfD2 = wf + 54784;  // KT=6,  MT=1 ->  3072
    _Float16* wfV0 = wf + 57856;  // 16x32  ->   512
    _Float16* wfV1 = wf + 58368;  // 32x32  ->  1024
    _Float16* wfV2 = wf + 59392;  // 64x64  ->  4096
    _Float16* wfV3 = wf + 63488;  // 128x128-> 16384

    // 1. Weight pre-swizzles to f16 fragment layouts.
    auto prep = [&](const float* w, _Float16* dst, int Cout, int Cin, int MT, int KT, int tr) {
        int total = KT * MT * 512;
        prep_weights_kernel<<<dim3((total + 255) / 256), dim3(256), 0, stream>>>(
            w, dst, Cout, Cin, MT, KT, tr);
    };
    prep(enc_w[0], wfE0, 12,  2, 1,  1, 0);
    prep(enc_w[1], wfE1, 24, 12, 2,  6, 0);
    prep(enc_w[2], wfE2, 48, 24, 3, 12, 0);
    prep(dec_w[0], wfD0, 24, 48, 2, 23, 1);
    prep(dec_w[1], wfD1, 12, 24, 1, 12, 1);
    prep(dec2_w,   wfD2,  2, 12, 1,  6, 1);

    static const int RW[4]  = { 16, 32, 64, 128 };
    static const int RWS[4] = { 32, 32, 64, 128 };
    _Float16* wfV[4] = { wfV0, wfV1, wfV2, wfV3 };
    for (int i = 0; i < 4; ++i) {
        int total = RW[i] * RWS[i];
        prep_deovw_kernel<<<dim3((total + 255) / 256), dim3(256), 0, stream>>>(
            deov_w[i], wfV[i], RW[i], RWS[i]);
    }

    // 2. OLA + interp + normalize.
    static const int RF[4]  = { 60, 60, 60, 30 };
    static const int RF0[4] = { 0, 60, 120, 180 };
    static const int RREP[4]= { 8, 4, 2, 1 };
    for (int i = 0; i < 4; ++i) {
        long long total = (long long)B_ * C_ * RF[i] * MAXT;
        ola_interp_kernel<<<dim3((unsigned)((total + 255) / 256)), dim3(256), 0, stream>>>(
            xin[i], mix, bufA, mean, scale, RF[i], RW[i], RF0[i], RREP[i]);
    }

    // 3. Encoder / decoder convs (ping-pong bufA <-> bufB, f16 activations).
    auto gx = [](int Wout) { return (unsigned)((Wout + WTILE - 1) / WTILE); };
    conv_wmma_kernel<1, 0><<<dim3(gx(8252), 178, 2), 128, 0, stream>>>(
        bufA, bufB, wfE0, enc_g[0], enc_b[0], enc_m[0], enc_v[0],
        2, 12, 180, 8256, 178, 8252, 1, 0, 0);
    conv_wmma_kernel<2, 0><<<dim3(gx(8248), 176, 2), 128, 0, stream>>>(
        bufB, bufA, wfE1, enc_g[1], enc_b[1], enc_m[1], enc_v[1],
        12, 24, 178, 8252, 176, 8248, 6, 0, 0);
    conv_wmma_kernel<3, 0><<<dim3(gx(8244), 174, 2), 128, 0, stream>>>(
        bufA, bufB, wfE2, enc_g[2], enc_b[2], enc_m[2], enc_v[2],
        24, 48, 176, 8248, 174, 8244, 12, 0, 0);
    conv_wmma_kernel<2, 0><<<dim3(gx(8248), 176, 2), 128, 0, stream>>>(
        bufB, bufA, wfD0, dec_g[0], dec_b[0], dec_m[0], dec_v[0],
        48, 24, 174, 8244, 176, 8248, 23, 2, 4);
    conv_wmma_kernel<1, 0><<<dim3(gx(8252), 178, 2), 128, 0, stream>>>(
        bufA, bufB, wfD1, dec_g[1], dec_b[1], dec_m[1], dec_v[1],
        24, 12, 176, 8248, 178, 8252, 12, 2, 4);
    conv_wmma_kernel<1, 2><<<dim3(gx(8256), 180, 2), 128, 0, stream>>>(
        bufB, bufA, wfD2, dec2_bias, dec2_bias, dec2_bias, dec2_bias,
        12, 2, 178, 8252, 180, 8256, 6, 2, 4);

    // 4. Mask + subsample + windowed einsum back to ragged outputs.
    deov_kernel<16><<<dim3(60, C_, B_), 128, 0, stream>>>(
        bufA, mix, mean, scale, wfV0, deov_b[0], out + 0,       60,   0, 8);
    deov_kernel<32><<<dim3(60, C_, B_), 128, 0, stream>>>(
        bufA, mix, mean, scale, wfV1, deov_b[1], out + 491520,  60,  60, 4);
    deov_kernel<64><<<dim3(60, C_, B_), 128, 0, stream>>>(
        bufA, mix, mean, scale, wfV2, deov_b[2], out + 1474560, 60, 120, 2);
    deov_kernel<128><<<dim3(30, C_, B_), 128, 0, stream>>>(
        bufA, mix, mean, scale, wfV3, deov_b[3], out + 3440640, 30, 180, 1);
}
